// RelationAwareMultiHeadAttention_45947560132771
// MI455X (gfx1250) — matrix operand order repaired and checked by reading, not compiled, not run on hardware
//
#include <hip/hip_runtime.h>

typedef __attribute__((ext_vector_type(16))) __bf16 v16bf;
typedef __attribute__((ext_vector_type(8)))  float  v8f;

constexpr int B_ = 2, S_ = 768, L_ = 767, H_ = 8, HD_ = 64, HID_ = 512, KCLIP_ = 64;
constexpr int RTB_ = S_ / 16;   // 48 row tiles per batch

union BF16x16 { v16bf v; unsigned short u[16]; uint4 q[2]; };

__device__ __forceinline__ unsigned short f2bfu(float f) {
  unsigned u = __float_as_uint(f);
  u += 0x7FFFu + ((u >> 16) & 1u);          // round-to-nearest-even
  return (unsigned short)(u >> 16);
}
__device__ __forceinline__ float bfu2f(unsigned h) {
  return __uint_as_float(h << 16);
}

// ---------------------------------------------------------------------------
// 0a) Elementwise f32 -> bf16 (vectorized), n must be a multiple of 4
// ---------------------------------------------------------------------------
__global__ void cvt_bf16_kernel(const float* __restrict__ in,
                                unsigned short* __restrict__ out, int n4)
{
  const int t = blockIdx.x * 256 + threadIdx.x;
  if (t >= n4) return;
  const float4 x = ((const float4*)in)[t];
  ushort4 o;
  o.x = f2bfu(x.x); o.y = f2bfu(x.y); o.z = f2bfu(x.z); o.w = f2bfu(x.w);
  ((ushort4*)out)[t] = o;
}

// ---------------------------------------------------------------------------
// 0b) Proj weight convert + transpose: W[H,512,64] f32 -> Wt[H,64,512] bf16
// ---------------------------------------------------------------------------
__global__ void wtrans_kernel(const float* __restrict__ W,
                              unsigned short* __restrict__ Wt)
{
  const int t = blockIdx.x * 256 + threadIdx.x;   // t = (h*64+d)*512 + k
  if (t >= H_ * HID_ * HD_) return;
  const int k  = t & (HID_ - 1);
  const int hd = t >> 9;
  const int d  = hd & (HD_ - 1), h = hd >> 6;
  Wt[t] = f2bfu(W[((size_t)h * HID_ + k) * HD_ + d]);
}

// ---------------------------------------------------------------------------
// 1) Row ranks of the pairwise-distance matrix (t1). t2 == t1^T by symmetry.
// ---------------------------------------------------------------------------
__global__ __launch_bounds__(256) void rank_rows_kernel(
    const float* __restrict__ coords,      // [B, L, 2]
    unsigned short* __restrict__ t1)       // [B, L, L]
{
  __shared__ float cx[L_ + 1], cy[L_ + 1], dist[L_ + 1];
  const int b = blockIdx.x / L_;
  const int i = blockIdx.x % L_;
  for (int j = threadIdx.x; j < L_; j += 256) {
    cx[j] = coords[((size_t)b * L_ + j) * 2 + 0];
    cy[j] = coords[((size_t)b * L_ + j) * 2 + 1];
  }
  __syncthreads();
  const float xi = cx[i], yi = cy[i];
  for (int j = threadIdx.x; j < L_; j += 256) {
    float dx = __fadd_rn(xi, -cx[j]);
    float dy = __fadd_rn(yi, -cy[j]);
    dist[j] = __fsqrt_rn(__fadd_rn(__fmul_rn(dx, dx), __fmul_rn(dy, dy)));
  }
  __syncthreads();
  for (int j = threadIdx.x; j < L_; j += 256) {
    const float dj = dist[j];
    int cnt = 0;
    for (int k = 0; k < L_; ++k) {
      float dk = dist[k];
      cnt += (dk < dj) || (dk == dj && k < j);
    }
    t1[((size_t)b * L_ + i) * L_ + j] = (unsigned short)cnt;
  }
}

// ---------------------------------------------------------------------------
// 2) idx = clip(triu(t1,1) + tril(t1^T,1), 0, 64), padded to [B,S,S] (u8)
//    jnp.triu(k=1) keeps j>=i+1, jnp.tril(k=1) keeps j<=i+1 -> the first
//    superdiagonal gets the SUM of both terms.
// ---------------------------------------------------------------------------
__global__ void build_idx_kernel(const unsigned short* __restrict__ t1,
                                 unsigned char* __restrict__ idx)
{
  const int t = blockIdx.x * 256 + threadIdx.x;
  if (t >= B_ * S_ * S_) return;
  const int b  = t / (S_ * S_);
  const int r  = t % (S_ * S_);
  const int qi = r / S_;
  const int kj = r % S_;
  int v = 0;
  if (qi > 0 && kj > 0) {
    const int i = qi - 1, j = kj - 1;
    if (j > i + 1)       v = t1[((size_t)b * L_ + i) * L_ + j];
    else if (j == i + 1) v = (int)t1[((size_t)b * L_ + i) * L_ + j] +
                             (int)t1[((size_t)b * L_ + j) * L_ + i];
    else                 v = t1[((size_t)b * L_ + j) * L_ + i];
    if (v > KCLIP_) v = KCLIP_;
  }
  idx[t] = (unsigned char)v;
}

// ---------------------------------------------------------------------------
// 3) Per-head projection GEMM via WMMA bf16 (all-bf16 operands, b128 loads).
//    grid (48, 8, 2), block 128 (4 waves; wave = 16-col tile of the head)
//    VT==0: out[b,h,s,d]  (Q,K);  VT==1: out[b,h,d,s]  (V transposed)
// ---------------------------------------------------------------------------
template <int VT>
__global__ __launch_bounds__(128) void proj_kernel(
    const unsigned short* __restrict__ Xb,  // [B*S, 512] bf16
    const unsigned short* __restrict__ Wt,  // [H, 64, 512] bf16
    const float* __restrict__ bias,         // [H, 64]
    unsigned short* __restrict__ out)       // bf16
{
  const int rt   = blockIdx.x;              // row tile within batch 0..47
  const int h    = blockIdx.y;              // head
  const int b    = blockIdx.z;              // batch
  const int wave = threadIdx.x >> 5, lane = threadIdx.x & 31;
  const int lhi  = (lane >= 16) ? 1 : 0, ln = lane & 15;
  const int row  = (b * RTB_ + rt) * 16 + ln;
  const int dcol = wave * 16 + ln;
  v8f c = {};
  for (int kb = 0; kb < 16; ++kb) {
    const int k0 = kb * 32;
    BF16x16 a, bm;
    const unsigned short* xp = Xb + (size_t)row * HID_ + k0 + lhi * 8;
    const unsigned short* wp =
        Wt + ((size_t)h * HD_ + dcol) * HID_ + k0 + lhi * 8;
    a.q[0]  = *(const uint4*)(xp);          // k = lhi*8 .. +7
    a.q[1]  = *(const uint4*)(xp + 16);     // k = 16+lhi*8 .. +7
    bm.q[0] = *(const uint4*)(wp);          // k = lhi*8 .. +7
    bm.q[1] = *(const uint4*)(wp + 16);
    c = __builtin_amdgcn_wmma_f32_16x16x32_bf16(false, a.v, false, bm.v,
                                                (short)0, c, false, false);
  }
  const float bb = bias[h * HD_ + dcol];
  const size_t hb = (size_t)(b * H_ + h);
#pragma unroll
  for (int v = 0; v < 8; ++v) {
    const int s = rt * 16 + v + lhi * 8;
    const size_t off = VT ? (hb * HD_ + dcol) * S_ + s
                          : (hb * S_ + s) * HD_ + dcol;
    out[off] = f2bfu(c[v] + bb);
  }
}

// ---------------------------------------------------------------------------
// 4) qpe[b,h,q,c] = dot(Q[b,h,q,:], pe_k[c,:])   c in 0..64
// ---------------------------------------------------------------------------
__global__ void qpe_kernel(const unsigned short* __restrict__ Qb,
                           const float* __restrict__ pe_k,
                           float* __restrict__ qpe)
{
  const int t = blockIdx.x * 256 + threadIdx.x;
  if (t >= B_ * H_ * S_ * 65) return;
  const int c  = t % 65;
  const int q  = (t / 65) % S_;
  const int bh = t / (65 * S_);
  const unsigned short* qp = Qb + ((size_t)bh * S_ + q) * HD_;
  const float* pk = pe_k + c * HD_;
  float acc = 0.f;
#pragma unroll
  for (int ch = 0; ch < 8; ++ch) {
    uint4 qv = *(const uint4*)(qp + ch * 8);
    const unsigned* pu = (const unsigned*)&qv;
#pragma unroll
    for (int w = 0; w < 4; ++w) {
      acc += bfu2f(pu[w] & 0xFFFFu) * pk[ch * 8 + w * 2];
      acc += bfu2f(pu[w] >> 16)     * pk[ch * 8 + w * 2 + 1];
    }
  }
  qpe[t] = acc;
}

// ---------------------------------------------------------------------------
// 5) Attention: one block per (b,h,16-query tile). 256 threads = 8 waves.
// ---------------------------------------------------------------------------
__global__ __launch_bounds__(256) void attn_kernel(
    const unsigned short* __restrict__ Qb,   // [B,H,S,64] bf16
    const unsigned short* __restrict__ Kb,   // [B,H,S,64] bf16
    const unsigned short* __restrict__ Vt,   // [B,H,64,S] bf16 (transposed)
    const float* __restrict__ qpe,           // [B,H,S,65]
    const unsigned char* __restrict__ idx,   // [B,S,S]
    const float* __restrict__ pe_v,          // [65,64]
    unsigned short* __restrict__ heads)      // [B,S,512] bf16
{
  __shared__ float sc[16][772];              // scores -> unnorm probs (f32)
  __shared__ float qpe_s[16][65];
  __shared__ float wsum[16][65];
  __shared__ float outacc[16][64];
  __shared__ float rowscale[16];

  const int qt = blockIdx.x;                 // 0..47
  const int bh = blockIdx.y;                 // 0..15
  const int b = bh >> 3, h = bh & 7;
  const int q0 = qt * 16;
  const int tid = threadIdx.x;
  const int wave = tid >> 5, lane = tid & 31;
  const int lhi = (lane >= 16) ? 1 : 0, ln = lane & 15;
  const size_t base = (size_t)bh * S_ * HD_;

  for (int t = tid; t < 16 * 65; t += 256) {
    const int m = t / 65, c = t % 65;
    wsum[m][c] = 0.f;
    qpe_s[m][c] = qpe[((size_t)bh * S_ + q0 + m) * 65 + c];
  }
  for (int t = tid; t < 16 * 64; t += 256) outacc[t >> 6][t & 63] = 0.f;
  __syncthreads();

  // ---- phase 1: scores = (Q K^T + qpe[idx]) / 8 ----
  BF16x16 aq0, aq1;
  {
    const unsigned short* qp = Qb + base + (size_t)(q0 + ln) * HD_ + lhi * 8;
    aq0.q[0] = *(const uint4*)(qp);          // k = lhi*8 .. +7
    aq0.q[1] = *(const uint4*)(qp + 16);     // k = 16+lhi*8 .. +7
    aq1.q[0] = *(const uint4*)(qp + 32);
    aq1.q[1] = *(const uint4*)(qp + 48);
  }
  for (int kt = wave; kt < 48; kt += 8) {
    const unsigned short* kp = Kb + base + (size_t)(kt * 16 + ln) * HD_ + lhi * 8;
    if (kt + 8 < 48) __builtin_prefetch(kp + 8 * 16 * HD_, 0, 3);
    BF16x16 bk0, bk1;
    bk0.q[0] = *(const uint4*)(kp);          // k = lhi*8 .. +7, 16+lhi*8 .. +7
    bk0.q[1] = *(const uint4*)(kp + 16);
    bk1.q[0] = *(const uint4*)(kp + 32);
    bk1.q[1] = *(const uint4*)(kp + 48);
    v8f c = {};
    c = __builtin_amdgcn_wmma_f32_16x16x32_bf16(false, aq0.v, false, bk0.v,
                                                (short)0, c, false, false);
    c = __builtin_amdgcn_wmma_f32_16x16x32_bf16(false, aq1.v, false, bk1.v,
                                                (short)0, c, false, false);
#pragma unroll
    for (int v = 0; v < 8; ++v) {
      const int m = v + lhi * 8;
      const int col = kt * 16 + ln;
      const int ci = idx[((size_t)b * S_ + q0 + m) * S_ + col];
      sc[m][col] = (c[v] + qpe_s[m][ci]) * 0.125f;
    }
  }
  __syncthreads();

  // ---- phase 2: softmax (unnormalized) + bucket scatter ----
  for (int r = 0; r < 2; ++r) {
    const int m = wave * 2 + r;
    float vmax = -3.4e38f;
    for (int col = lane; col < S_; col += 32) vmax = fmaxf(vmax, sc[m][col]);
#pragma unroll
    for (int off = 16; off >= 1; off >>= 1)
      vmax = fmaxf(vmax, __shfl_xor(vmax, off, 32));
    float psum = 0.f;
    const unsigned char* ip = idx + ((size_t)b * S_ + q0 + m) * S_;
    for (int col = lane; col < S_; col += 32) {
      const float p = __expf(sc[m][col] - vmax);
      sc[m][col] = p;
      psum += p;
      atomicAdd(&wsum[m][ip[col]], p);
    }
#pragma unroll
    for (int off = 16; off >= 1; off >>= 1) psum += __shfl_xor(psum, off, 32);
    if (lane == 0) rowscale[m] = 1.0f / psum;
  }
  __syncthreads();

  // ---- phase 3: P @ V (each wave owns a contiguous 96-key slab) ----
  const v8f zero = {};
  v8f oacc[4];
#pragma unroll
  for (int nt = 0; nt < 4; ++nt) oacc[nt] = zero;
  for (int ch = 0; ch < 3; ++ch) {
    const int kbase = wave * 96 + ch * 32;
    BF16x16 ap;
#pragma unroll
    for (int e = 0; e < 16; ++e) {
      const int k = kbase + e + ((e >= 8) ? 8 : 0) + (lhi ? 8 : 0);
      ap.u[e] = f2bfu(sc[ln][k]);
    }
#pragma unroll
    for (int nt = 0; nt < 4; ++nt) {
      // V^T row (d fixed) is contiguous along keys -> 2x b128 loads
      const unsigned short* vp =
          Vt + (base + (size_t)(nt * 16 + ln) * S_) + kbase + lhi * 8;
      if (ch + 1 < 3) __builtin_prefetch(vp + 32, 0, 3);
      BF16x16 bv;
      bv.q[0] = *(const uint4*)(vp);
      bv.q[1] = *(const uint4*)(vp + 16);
      oacc[nt] = __builtin_amdgcn_wmma_f32_16x16x32_bf16(
          false, ap.v, false, bv.v, (short)0, oacc[nt], false, false);
    }
  }
#pragma unroll
  for (int nt = 0; nt < 4; ++nt)
#pragma unroll
    for (int v = 0; v < 8; ++v)
      atomicAdd(&outacc[v + lhi * 8][nt * 16 + ln], oacc[nt][v]);
  __syncthreads();

  // ---- phase 4: add wsum @ pe_v, normalize, write heads (bf16) ----
  for (int t = tid; t < 16 * 64; t += 256) {
    const int m = t >> 6, d = t & 63;
    float acc = outacc[m][d];
    for (int c = 0; c <= 64; ++c) acc += wsum[m][c] * pe_v[c * HD_ + d];
    heads[((size_t)b * S_ + q0 + m) * HID_ + h * HD_ + d] =
        f2bfu(acc * rowscale[m]);
  }
}

// ---------------------------------------------------------------------------
// 6) Output projection: heads @ fc_W^T + fc_b  (all-bf16 WMMA, b128 loads)
//    grid (96, 8), block 128 (4 waves; wave owns one 16-col tile)
// ---------------------------------------------------------------------------
__global__ __launch_bounds__(128) void fc_kernel(
    const unsigned short* __restrict__ heads,   // [B*S, 512] bf16
    const unsigned short* __restrict__ Wfb,     // [512, 512] bf16 (row = o)
    const float* __restrict__ bf,               // [512]
    float* __restrict__ out)                    // [B*S, 512] f32
{
  const int rt = blockIdx.x;
  const int wave = threadIdx.x >> 5, lane = threadIdx.x & 31;
  const int ct = blockIdx.y * 4 + wave;
  const int lhi = (lane >= 16) ? 1 : 0, ln = lane & 15;
  const int row = rt * 16 + ln, o = ct * 16 + ln;
  v8f c = {};
  for (int kb = 0; kb < 16; ++kb) {
    const int k0 = kb * 32;
    BF16x16 a, bm;
    const unsigned short* hp = heads + (size_t)row * HID_ + k0 + lhi * 8;
    const unsigned short* wp = Wfb + (size_t)o * HID_ + k0 + lhi * 8;
    a.q[0]  = *(const uint4*)(hp);
    a.q[1]  = *(const uint4*)(hp + 16);
    bm.q[0] = *(const uint4*)(wp);
    bm.q[1] = *(const uint4*)(wp + 16);
    c = __builtin_amdgcn_wmma_f32_16x16x32_bf16(false, a.v, false, bm.v,
                                                (short)0, c, false, false);
  }
  const float bb = bf[o];
#pragma unroll
  for (int v = 0; v < 8; ++v)
    out[((size_t)(rt * 16 + v + lhi * 8)) * HID_ + o] = c[v] + bb;
}

// ---------------------------------------------------------------------------
extern "C" void kernel_launch(void* const* d_in, const int* in_sizes, int n_in,
                              void* d_out, int out_size, void* d_ws, size_t ws_size,
                              hipStream_t stream) {
  (void)in_sizes; (void)n_in; (void)out_size; (void)ws_size;

  const float* query  = (const float*)d_in[0];
  const float* key    = (const float*)d_in[1];
  const float* value  = (const float*)d_in[2];
  const float* coords = (const float*)d_in[3];
  const float* Wq = (const float*)d_in[4];
  const float* bq = (const float*)d_in[5];
  const float* Wk = (const float*)d_in[6];
  const float* bk = (const float*)d_in[7];
  const float* Wv = (const float*)d_in[8];
  const float* bv = (const float*)d_in[9];
  const float* pe_k = (const float*)d_in[10];
  const float* pe_v = (const float*)d_in[11];
  const float* fc_W = (const float*)d_in[12];
  const float* fc_b = (const float*)d_in[13];
  float* out = (float*)d_out;

  // workspace carve-up (256B aligned)
  char* w = (char*)d_ws;
  auto carve = [&](size_t bytes) -> void* {
    void* p = (void*)w;
    w += (bytes + 255) & ~(size_t)255;
    return p;
  };
  const size_t nX  = (size_t)B_ * S_ * HID_;       // 786432
  const size_t nW  = (size_t)H_ * HID_ * HD_;      // 262144
  const size_t nQ  = (size_t)B_ * H_ * S_ * HD_;   // 786432
  unsigned short* t1    = (unsigned short*)carve((size_t)B_ * L_ * L_ * 2);
  unsigned char*  idx   = (unsigned char*) carve((size_t)B_ * S_ * S_);
  unsigned short* Xqb   = (unsigned short*)carve(nX * 2);
  unsigned short* Xkb   = (unsigned short*)carve(nX * 2);
  unsigned short* Xvb   = (unsigned short*)carve(nX * 2);
  unsigned short* Wqt   = (unsigned short*)carve(nW * 2);
  unsigned short* Wkt   = (unsigned short*)carve(nW * 2);
  unsigned short* Wvt   = (unsigned short*)carve(nW * 2);
  unsigned short* Wfb   = (unsigned short*)carve((size_t)HID_ * HID_ * 2);
  unsigned short* Qb    = (unsigned short*)carve(nQ * 2);
  unsigned short* Kb    = (unsigned short*)carve(nQ * 2);
  unsigned short* Vt    = (unsigned short*)carve(nQ * 2);
  float*          qpe   = (float*)         carve((size_t)B_ * H_ * S_ * 65 * 4);
  unsigned short* heads = (unsigned short*)carve(nX * 2);

  // precision pre-passes (one-time converts / transposes)
  cvt_bf16_kernel<<<(int)(nX / 4 + 255) / 256, 256, 0, stream>>>(query, Xqb, nX / 4);
  cvt_bf16_kernel<<<(int)(nX / 4 + 255) / 256, 256, 0, stream>>>(key,   Xkb, nX / 4);
  cvt_bf16_kernel<<<(int)(nX / 4 + 255) / 256, 256, 0, stream>>>(value, Xvb, nX / 4);
  cvt_bf16_kernel<<<(int)(HID_ * HID_ / 4 + 255) / 256, 256, 0, stream>>>(
      fc_W, Wfb, HID_ * HID_ / 4);
  wtrans_kernel<<<(int)(nW + 255) / 256, 256, 0, stream>>>(Wq, Wqt);
  wtrans_kernel<<<(int)(nW + 255) / 256, 256, 0, stream>>>(Wk, Wkt);
  wtrans_kernel<<<(int)(nW + 255) / 256, 256, 0, stream>>>(Wv, Wvt);

  rank_rows_kernel<<<dim3(B_ * L_), 256, 0, stream>>>(coords, t1);
  build_idx_kernel<<<(B_ * S_ * S_ + 255) / 256, 256, 0, stream>>>(t1, idx);

  proj_kernel<0><<<dim3(RTB_, H_, B_), 128, 0, stream>>>(Xqb, Wqt, bq, Qb);
  proj_kernel<0><<<dim3(RTB_, H_, B_), 128, 0, stream>>>(Xkb, Wkt, bk, Kb);
  proj_kernel<1><<<dim3(RTB_, H_, B_), 128, 0, stream>>>(Xvb, Wvt, bv, Vt);

  qpe_kernel<<<(B_ * H_ * S_ * 65 + 255) / 256, 256, 0, stream>>>(Qb, pe_k, qpe);

  attn_kernel<<<dim3(S_ / 16, B_ * H_), 256, 0, stream>>>(Qb, Kb, Vt, qpe, idx,
                                                          pe_v, heads);

  fc_kernel<<<dim3(B_ * S_ / 16, HID_ / 16 / 4), 128, 0, stream>>>(heads, Wfb,
                                                                   fc_b, out);
}
